// ScalingSelfAtt3Dot_65094524338379
// MI455X (gfx1250) — compile-verified, matrix operand order checked by reading, not run on hardware
//
#include <hip/hip_runtime.h>

// ---------------------------------------------------------------------------
// CDNA5 (gfx1250, wave32) fused ScalingSelfAtt3D.
//  k1: init reduction workspace
//  k2: fused [async-stage x -> q,k,v WMMA f16 (weights staged once in LDS) ->
//      per-head softmax -> SE(32->16->64 via WMMA) -> sigmoid -> store +
//      per-channel max/sum/min atomics, plus sup_q1/sup_q2 thin convs]
//  k3: gate (pool -> fc -> relu -> fc -> sigmoid), 1 block
//  k4: scale o1/o2 in place by gates (streaming, float4)
// ---------------------------------------------------------------------------

#if defined(__has_builtin)
#  if __has_builtin(__builtin_amdgcn_global_load_async_to_lds_b128) && \
      __has_builtin(__builtin_amdgcn_s_wait_asynccnt)
#    define HAVE_ASYNC_LDS 1
#  endif
#endif
#ifndef HAVE_ASYNC_LDS
#  define HAVE_ASYNC_LDS 0
#endif

typedef __attribute__((ext_vector_type(16))) _Float16 v16h;
typedef __attribute__((ext_vector_type(8)))  float    v8f;
typedef int v4i_ __attribute__((vector_size(16)));    // matches builtin param type

union AFrag { v16h h; unsigned u[8]; };
union H2    { unsigned u; _Float16 h[2]; };

static __device__ __forceinline__ v8f wmma16(v16h a, v16h b, v8f c) {
  // v_wmma_f32_16x16x32_f16
  return __builtin_amdgcn_wmma_f32_16x16x32_f16(false, a, false, b,
                                                (short)0, c, false, false);
}

// 16x32 f16 fragment from a row-major f16 LDS tile [row][K], rowstride halves.
// 16-bit layout: lanes 0-15 -> row=lane, K={0..7,16..23}; lanes 16-31 -> K+8.
// Used both for A (rows = positions) and B (rows = output channels).
static __device__ __forceinline__ v16h frag_lds(const _Float16* base, int lane,
                                                int r0, int rowstride, int kbase) {
  const int m = lane & 15, hi = (lane >> 4) & 1;
  const _Float16* row = base + (r0 + m) * rowstride + kbase + hi * 8;
  AFrag f;
#pragma unroll
  for (int pr = 0; pr < 8; ++pr) {
    const int k = ((pr & 4) ? 16 : 0) + (pr & 3) * 2;
    f.u[pr] = *(const unsigned*)(row + k);   // ds_load_b32 (half2)
  }
  return f.h;
}

// A fragment from channel-major f32 LDS x-tile xs[64 ch][64 pos].
static __device__ __forceinline__ v16h afrag_xf(const float* xs, int lane,
                                                int p0, int kbase) {
  const int m = lane & 15, hi = (lane >> 4) & 1;
  const int pos = p0 + m;
  AFrag f;
#pragma unroll
  for (int pr = 0; pr < 8; ++pr) {
    const int k = kbase + ((pr & 4) ? 16 : 0) + (pr & 3) * 2 + hi * 8;
    H2 p;
    p.h[0] = (_Float16)xs[k * 64 + pos];
    p.h[1] = (_Float16)xs[(k + 1) * 64 + pos];
    f.u[pr] = p.u;
  }
  return f.h;
}

// SE layer-2 B fragment from LDS f16 W2[64][16]; valid K band [hf*16,hf*16+16).
static __device__ __forceinline__ v16h bfrag_w2(const _Float16* W2, int lane,
                                                int n0, int hf) {
  const int n = lane & 15, hi = (lane >> 4) & 1;
  const _Float16* row = W2 + (n0 + n) * 16;
  AFrag f;
#pragma unroll
  for (int pr = 0; pr < 8; ++pr) {
    const int k = ((pr & 4) ? 16 : 0) + (pr & 3) * 2 + hi * 8;
    const int lo = k - hf * 16;
    if (lo >= 0 && lo < 16) f.u[pr] = *(const unsigned*)(row + lo);
    else                    f.u[pr] = 0u;
  }
  return f.h;
}

#define NPB      (1 << 17)          // 131072 spatial positions per batch
#define O2_OFF   16777216u
#define SQ1_OFF  33554432u
#define SQ2_OFF  35127296u

__global__ void init_ws(float* wsum, unsigned* wmax, unsigned* wmin) {
  const int t = blockIdx.x * blockDim.x + threadIdx.x;
  if (t < 256) { wsum[t] = 0.0f; wmax[t] = 0u; wmin[t] = 0x7f800000u; }
}

// xs (f32 x-tile, channel-major) and oss (sigmoid outputs, position-major)
// share storage: both use 256B rows, and
// live ranges are separated (xs consumed right after the post-stage barrier).
union XOu {
  float    xs[64][64];     // [ch][pos]
  _Float16 oss[64][128];   // [pos][o1(64)|o2(64)]
};

__global__ __launch_bounds__(128) void fused_main(
    const float* __restrict__ x,
    const float* __restrict__ Wq,  const float* __restrict__ bq,
    const float* __restrict__ Wk,  const float* __restrict__ bk,
    const float* __restrict__ Wv,  const float* __restrict__ bv,
    const float* __restrict__ Wq1, const float* __restrict__ bq1,
    const float* __restrict__ Wq2, const float* __restrict__ bq2,
    const float* __restrict__ se1w1, const float* __restrict__ se1b1,
    const float* __restrict__ se1w2, const float* __restrict__ se1b2,
    const float* __restrict__ se2w1, const float* __restrict__ se2b1,
    const float* __restrict__ se2w2, const float* __restrict__ se2b2,
    float* __restrict__ out,
    float* __restrict__ wsum, unsigned* __restrict__ wmax, unsigned* __restrict__ wmin)
{
  __shared__ _Float16 wqL[64][64];     // staged W (f16, [o][c])   8KB each
  __shared__ _Float16 wkL[64][64];
  __shared__ _Float16 wvL[64][64];
  __shared__ _Float16 sw1L[2][16][32]; // SE layer-1 weights       2KB
  __shared__ _Float16 sw2L[2][64][16]; // SE layer-2 weights       4KB
  __shared__ XOu      xo;              // x-tile / sigmoid out    16KB
  __shared__ _Float16 qs[64][64];      // q tile for sup convs     8KB
  __shared__ _Float16 od[64][64];      // attn out; cols 0..31 reused as SE hidden

  const int tid  = threadIdx.x;
  const int wave = tid >> 5, lane = tid & 31;
  const int n    = lane & 15, hi = (lane >> 4) & 1;
  const int p0   = wave * 16;          // each wave owns 16 positions

  // ---- stage all weights once per block (f32 -> f16) ----
  for (int i = tid; i < 64 * 64; i += 128) {
    const int r = i >> 6, c = i & 63;
    wqL[r][c] = (_Float16)Wq[i];
    wkL[r][c] = (_Float16)Wk[i];
    wvL[r][c] = (_Float16)Wv[i];
  }
  for (int i = tid; i < 16 * 32; i += 128) {
    sw1L[0][i >> 5][i & 31] = (_Float16)se1w1[i];
    sw1L[1][i >> 5][i & 31] = (_Float16)se2w1[i];
  }
  for (int i = tid; i < 64 * 16; i += 128) {
    sw2L[0][i >> 4][i & 15] = (_Float16)se1w2[i];
    sw2L[1][i >> 4][i & 15] = (_Float16)se2w2[i];
  }
  __syncthreads();

  for (int t = 0; t < 8; ++t) {
    const int tile = blockIdx.x * 8 + t;
    const int pg   = tile * 64;
    const int b    = pg >> 17;
    const int s0   = pg & (NPB - 1);

    // ---- stage x tile: global f32 -> LDS [ch][pos] (async if available) ----
#if HAVE_ASYNC_LDS
    for (int i = tid; i < 1024; i += 128) {
      const int c = i >> 4, p4 = (i & 15) << 2;
      const size_t gi = ((size_t)(b * 64 + c) << 17) + (size_t)(s0 + p4);
      __builtin_amdgcn_global_load_async_to_lds_b128(
          (__attribute__((address_space(1))) v4i_*)(x + gi),
          (__attribute__((address_space(3))) v4i_*)(&xo.xs[c][p4]), 0, 0);
    }
    __builtin_amdgcn_s_wait_asynccnt(0);
#else
    for (int i = tid; i < 1024; i += 128) {
      const int c = i >> 4, p4 = (i & 15) << 2;
      const size_t gi = ((size_t)(b * 64 + c) << 17) + (size_t)(s0 + p4);
      *(float4*)&xo.xs[c][p4] = *(const float4*)(x + gi);
      __builtin_prefetch(&x[gi + 64], 0, 1);   // next tile (global_prefetch)
    }
#endif
    __syncthreads();

    const v16h A0 = afrag_xf(&xo.xs[0][0], lane, p0, 0);
    const v16h A1 = afrag_xf(&xo.xs[0][0], lane, p0, 32);
    __syncthreads();   // x-tile fully consumed block-wide (xs aliases oss)

    // ---- q,k,v GEMMs (weights from LDS) + per-head softmax attention ----
    for (int nt = 0; nt < 4; ++nt) {
      v8f qa = {}, ka = {}, va = {};
      qa = wmma16(A0, frag_lds(&wqL[0][0], lane, nt * 16, 64, 0),  qa);
      qa = wmma16(A1, frag_lds(&wqL[0][0], lane, nt * 16, 64, 32), qa);
      ka = wmma16(A0, frag_lds(&wkL[0][0], lane, nt * 16, 64, 0),  ka);
      ka = wmma16(A1, frag_lds(&wkL[0][0], lane, nt * 16, 64, 32), ka);
      va = wmma16(A0, frag_lds(&wvL[0][0], lane, nt * 16, 64, 0),  va);
      va = wmma16(A1, frag_lds(&wvL[0][0], lane, nt * 16, 64, 32), va);
      const int ch = nt * 16 + n;
      const float bqv = bq[ch], bkv = bk[ch], bvv = bv[ch];
#pragma unroll
      for (int r = 0; r < 8; ++r) {
        const int m = p0 + r + hi * 8;
        const float qf = qa[r] + bqv;
        const float kf = ka[r] + bkv;
        const float vf = va[r] + bvv;
        qs[m][ch] = (_Float16)qf;
        // softmax over the 4 lanes of one head (scale = 4^-0.5 = 0.5)
        float s  = qf * kf * 0.5f;
        float mx = fmaxf(s, __shfl_xor(s, 1));
        mx       = fmaxf(mx, __shfl_xor(mx, 2));
        const float e = __expf(s - mx);
        float sm = e + __shfl_xor(e, 1);
        sm      += __shfl_xor(sm, 2);
        od[m][ch] = (_Float16)((e / sm) * vf);
      }
    }
    // (no barrier: od/qs rows are wave-local; DS ops are in-order per wave)

    // ---- SE layer 1: 32 -> 16, relu; hidden written back into od cols 0..31
    {
      const v16h Ao1 = frag_lds(&od[0][0], lane, p0, 64, 0);   // out1 (K=32)
      const v16h Ao2 = frag_lds(&od[0][0], lane, p0, 64, 32);  // out2 (K=32)
      v8f h1 = {}, h2 = {};
      h1 = wmma16(Ao1, frag_lds(&sw1L[0][0][0], lane, 0, 32, 0), h1);
      h2 = wmma16(Ao2, frag_lds(&sw1L[1][0][0], lane, 0, 32, 0), h2);
      const float b1a = se1b1[n], b1b = se2b1[n];
#pragma unroll
      for (int r = 0; r < 8; ++r) {
        const int m = p0 + r + hi * 8;
        od[m][n]      = (_Float16)fmaxf(h1[r] + b1a, 0.0f);  // hidden1 -> K 0..15
        od[m][16 + n] = (_Float16)fmaxf(h2[r] + b1b, 0.0f);  // hidden2 -> K 16..31
      }
    }

    // ---- SE layer 2: 16 -> 64, sigmoid (both halves share one A frag) ----
    {
      const v16h Ah = frag_lds(&od[0][0], lane, p0, 64, 0);
      for (int hf = 0; hf < 2; ++hf) {
        const _Float16* w2 = &sw2L[hf][0][0];
        const float*    b2 = hf ? se2b2 : se1b2;
        for (int nt = 0; nt < 4; ++nt) {
          v8f a = {};
          a = wmma16(Ah, bfrag_w2(w2, lane, nt * 16, hf), a);
          const float bb = b2[nt * 16 + n];
#pragma unroll
          for (int r = 0; r < 8; ++r) {
            const int m = p0 + r + hi * 8;
            const float sv = 1.0f / (1.0f + __expf(-(a[r] + bb)));
            xo.oss[m][hf * 64 + nt * 16 + n] = (_Float16)sv;  // wave-local rows
          }
        }
      }
    }
    __syncthreads();

    // ---- sup_q1 (6 ch from q[:,:32]) and sup_q2 (1 ch from q[:,32:]) ----
    for (int j = tid; j < 64 * 7; j += 128) {
      const int p  = j / 7, o = j % 7;
      const int cb = (o == 6) ? 32 : 0;
      const float* wr = (o == 6) ? Wq2 : (Wq1 + o * 32);
      float acc = (o == 6) ? bq2[0] : bq1[o];
      for (int c = 0; c < 32; ++c) acc += (float)qs[p][cb + c] * wr[c];
      const size_t off = (o == 6)
          ? (size_t)SQ2_OFF + ((size_t)b << 17) + (size_t)(s0 + p)
          : (size_t)SQ1_OFF + ((size_t)(b * 6 + o) << 17) + (size_t)(s0 + p);
      out[off] = acc;
    }

    // ---- store o1_s/o2_s + per-(b,channel) max/sum/min atomics ----
    {
      const int ch128 = tid;                  // 128 threads = 128 channels
      const int hf = ch128 >> 6, ch = ch128 & 63;
      float mx = 0.0f, mn = __uint_as_float(0x7f800000u), sm = 0.0f;
      const size_t base = (size_t)hf * O2_OFF +
                          ((size_t)(b * 64 + ch) << 17) + (size_t)s0;
      for (int p = 0; p < 64; ++p) {
        const float v = (float)xo.oss[p][ch128];
        out[base + p] = v;
        mx = fmaxf(mx, v); mn = fminf(mn, v); sm += v;
      }
      const int ridx = b * 128 + ch128;
      atomicAdd(&wsum[ridx], sm);
      atomicMax(&wmax[ridx], __float_as_uint(mx));  // sigmoid > 0 -> uint order ok
      atomicMin(&wmin[ridx], __float_as_uint(mn));
    }
    __syncthreads();
  }
}

__global__ __launch_bounds__(256) void gate_kernel(
    const float* __restrict__ sq1w, const float* __restrict__ sq1b,
    const float* __restrict__ sq2w, const float* __restrict__ sq2b,
    const float* __restrict__ c1w1, const float* __restrict__ c1b1,
    const float* __restrict__ c1w2, const float* __restrict__ c1b2,
    const float* __restrict__ c2w1, const float* __restrict__ c2b1,
    const float* __restrict__ c2w2, const float* __restrict__ c2b2,
    const float* __restrict__ wsum, const unsigned* __restrict__ wmax,
    const unsigned* __restrict__ wmin, float* __restrict__ g)
{
  __shared__ float pl[2][128];
  __shared__ float hid[2][2][16];
  const int t = threadIdx.x;
  const int b = t >> 7, ch128 = t & 127, hf = ch128 >> 6;
  {
    const float mx = __uint_as_float(wmax[t]);
    const float mn = __uint_as_float(wmin[t]);
    const float av = wsum[t] * (1.0f / 131072.0f);
    const float* sw = hf ? sq2w : sq1w;
    const float sb  = hf ? sq2b[0] : sq1b[0];
    pl[b][ch128] = mx * sw[0] + av * sw[1] + mn * sw[2] + sb;
  }
  __syncthreads();
  if (t < 64) {
    const int bb = t >> 5, hh = (t >> 4) & 1, h = t & 15;
    const float* w1 = hh ? c2w1 : c1w1;
    float acc = (hh ? c2b1 : c1b1)[h];
    for (int c = 0; c < 64; ++c) acc += pl[bb][hh * 64 + c] * w1[h * 64 + c];
    hid[bb][hh][h] = fmaxf(acc, 0.0f);
  }
  __syncthreads();
  {
    const int c = ch128 & 63;
    const float* w2 = hf ? c2w2 : c1w2;
    float acc = (hf ? c2b2 : c1b2)[c];
    for (int h = 0; h < 16; ++h) acc += hid[b][hf][h] * w2[c * 16 + h];
    g[t] = 1.0f / (1.0f + __expf(-acc));
  }
}

__global__ __launch_bounds__(256) void scale_kernel(float* __restrict__ out,
                                                    const float* __restrict__ g) {
  const long long stride = (long long)gridDim.x * blockDim.x;
  for (long long i = (long long)blockIdx.x * blockDim.x + threadIdx.x;
       i < 8388608LL; i += stride) {
    const long long idx = i * 4;                 // float4-aligned, ch-uniform
    const int hf = (int)(idx >> 24);
    const int b  = (int)((idx >> 23) & 1);
    const int ch = (int)((idx >> 17) & 63);
    const float gg = g[b * 128 + hf * 64 + ch];
    float4* p = (float4*)(out + idx);
    float4 v = *p;
    v.x *= gg; v.y *= gg; v.z *= gg; v.w *= gg;
    *p = v;
  }
}

extern "C" void kernel_launch(void* const* d_in, const int* in_sizes, int n_in,
                              void* d_out, int out_size, void* d_ws, size_t ws_size,
                              hipStream_t stream) {
  (void)in_sizes; (void)n_in; (void)out_size; (void)ws_size;
  const float* x     = (const float*)d_in[0];
  const float* Wq    = (const float*)d_in[1];
  const float* bq    = (const float*)d_in[2];
  const float* Wk    = (const float*)d_in[3];
  const float* bk    = (const float*)d_in[4];
  const float* Wv    = (const float*)d_in[5];
  const float* bv    = (const float*)d_in[6];
  const float* Wq1   = (const float*)d_in[7];
  const float* bq1   = (const float*)d_in[8];
  const float* Wq2   = (const float*)d_in[9];
  const float* bq2   = (const float*)d_in[10];
  const float* se1w1 = (const float*)d_in[11];
  const float* se1b1 = (const float*)d_in[12];
  const float* se1w2 = (const float*)d_in[13];
  const float* se1b2 = (const float*)d_in[14];
  const float* se2w1 = (const float*)d_in[15];
  const float* se2b1 = (const float*)d_in[16];
  const float* se2w2 = (const float*)d_in[17];
  const float* se2b2 = (const float*)d_in[18];
  const float* sq1w  = (const float*)d_in[19];
  const float* sq1b  = (const float*)d_in[20];
  const float* sq2w  = (const float*)d_in[21];
  const float* sq2b  = (const float*)d_in[22];
  const float* c1w1  = (const float*)d_in[23];
  const float* c1b1  = (const float*)d_in[24];
  const float* c1w2  = (const float*)d_in[25];
  const float* c1b2  = (const float*)d_in[26];
  const float* c2w1  = (const float*)d_in[27];
  const float* c2b1  = (const float*)d_in[28];
  const float* c2w2  = (const float*)d_in[29];
  const float* c2b2  = (const float*)d_in[30];

  float*    out  = (float*)d_out;
  float*    wsf  = (float*)d_ws;
  float*    wsum = wsf;                         // [256]
  unsigned* wmax = (unsigned*)(wsf + 256);      // [256]
  unsigned* wmin = (unsigned*)(wsf + 512);      // [256]
  float*    g    = wsf + 768;                   // [256]

  init_ws<<<1, 256, 0, stream>>>(wsum, wmax, wmin);
  fused_main<<<512, 128, 0, stream>>>(x, Wq, bq, Wk, bk, Wv, bv,
                                      Wq1, bq1, Wq2, bq2,
                                      se1w1, se1b1, se1w2, se1b2,
                                      se2w1, se2b1, se2w2, se2b2,
                                      out, wsum, wmax, wmin);
  gate_kernel<<<1, 256, 0, stream>>>(sq1w, sq1b, sq2w, sq2b,
                                     c1w1, c1b1, c1w2, c1b2,
                                     c2w1, c2b1, c2w2, c2b2,
                                     wsum, wmax, wmin, g);
  scale_kernel<<<2048, 256, 0, stream>>>(out, g);
}
